// mask_star_satelitte_planet_pretraining_model_35923106464138
// MI455X (gfx1250) — matrix operand order repaired and checked by reading, not compiled, not run on hardware
//
#include <hip/hip_runtime.h>
#include <hip/hip_bf16.h>
#include <stddef.h>
#include <stdint.h>

// ---------------- model constants ----------------
constexpr int BB = 8;          // batch
constexpr int LL = 1024;       // seq len
constexpr int HH = 768;        // hidden (= D)
constexpr int NHEAD = 12;
constexpr int HDIM = 64;
constexpr int DINNER = 3072;
constexpr int NLAYERS = 2;
constexpr int SSTAR = LL + 1;  // 1025 keys for star attention
constexpr int NHID = 1024;
constexpr int NOUT = 5000;
constexpr float SCALE_ATT = 0.125f;   // 1/sqrt(64)
constexpr float LEAKY_SLOPE = 0.01f;

// ---------------- vector types for WMMA ----------------
typedef __attribute__((ext_vector_type(16))) __bf16          bf16x16;
typedef __attribute__((ext_vector_type(8)))  float           f32x8;
typedef __attribute__((ext_vector_type(8)))  unsigned short  u16x8;
typedef __attribute__((ext_vector_type(16))) unsigned short  u16x16;
typedef __attribute__((ext_vector_type(4)))  int             v4i;

// fp32 -> bf16 (round-to-nearest-even), raw u16 storage
__device__ __host__ inline unsigned short f2bf(float f) {
  union { float f; unsigned u; } x; x.f = f;
  unsigned r = x.u + 0x7FFFu + ((x.u >> 16) & 1u);
  return (unsigned short)(r >> 16);
}

__device__ inline void storeC(float* p, float v) { *p = v; }
__device__ inline void storeC(unsigned short* p, float v) { *p = f2bf(v); }

// ---------------- CDNA5 async-to-LDS (guarded; falls back cleanly) ----------------
#if defined(__HIP_DEVICE_COMPILE__) && __has_builtin(__builtin_amdgcn_global_load_async_to_lds_b128)
#define USE_ASYNC_LDS 1
#else
#define USE_ASYNC_LDS 0
#endif

#if USE_ASYNC_LDS
// prototype (from hipcc diagnostic): first arg is v4i32* in AS1 ("__device__"),
// LDS-side dst in AS3, then imm offset + imm cpol.
__device__ inline void async_copy16(const void* g, void* l) {
  __builtin_amdgcn_global_load_async_to_lds_b128(
      (__attribute__((address_space(1))) v4i*)(g),
      (__attribute__((address_space(3))) v4i*)(l), 0, 0);
}
#endif

template<int N>
__device__ inline void s_wait_async() {
#if __has_builtin(__builtin_amdgcn_s_wait_asynccnt)
  __builtin_amdgcn_s_wait_asynccnt(N);
#else
  asm volatile("s_wait_asynccnt %0" :: "i"(N) : "memory");
#endif
}

// =====================================================================
// GEMM: C[M,N] = act(A[M,K](bf16,row-major) * W[N,K](bf16,row-major)^T + bias)
// block = 256 threads (8 waves), tile 128x128, K-step 32, double-buffered LDS,
// wave tile 32x64 -> 8x v_wmma_f32_16x16x32_bf16 per wave per K-step.
// Out-of-range rows/cols: addresses clamped (garbage lands only in rows/cols
// the bounds-checked epilogue never stores).
// =====================================================================
template<bool RELU, typename OutT>
__global__ __launch_bounds__(256)
void gemm_bf16_wmma(const unsigned short* __restrict__ A,
                    const unsigned short* __restrict__ W,
                    const float* __restrict__ bias,
                    OutT* __restrict__ C,
                    int M, int N, int K, int ldc)
{
  constexpr int BM = 128, BN = 128, BK = 32;
  constexpr int LDS_S = BK + 8;                 // 40 elems = 80B rows (16B aligned)
  __shared__ unsigned short As[2][BM * LDS_S];
  __shared__ unsigned short Ws[2][BN * LDS_S];

  const int tid  = threadIdx.x;
  const int lane = tid & 31;
  const int wid  = tid >> 5;
  const int wm   = wid >> 1;        // 0..3 -> 32 rows each
  const int wn   = wid & 1;         // 0..1 -> 64 cols each
  const int m0 = blockIdx.y * BM;
  const int n0 = blockIdx.x * BN;

  // staging coordinates: 2 chunks (16B) each for A and W per thread
  int srow[2], skc[2];
  const unsigned short* aptr[2];
  const unsigned short* wptr[2];
  #pragma unroll
  for (int i = 0; i < 2; ++i) {
    const int c = tid + i * 256;
    srow[i] = c >> 2;
    skc[i]  = (c & 3) * 8;
    int ga = m0 + srow[i]; ga = (ga < M) ? ga : (M - 1);
    int gw = n0 + srow[i]; gw = (gw < N) ? gw : (N - 1);
    aptr[i] = A + (size_t)ga * K + skc[i];
    wptr[i] = W + (size_t)gw * K + skc[i];
  }

  f32x8 acc[2][4] = {};
  const int T = K / BK;   // K is always a multiple of 32 here

  // fragment addressing (ISA 16x32 A / 32x16 B bf16 layouts)
  const int lane15 = lane & 15;
  const int kA = (lane < 16) ? 0 : 8;
  const int kB = (lane < 16) ? 0 : 8;

  auto compute = [&](int p) {
    u16x16 afrag[2], bfrag[4];
    #pragma unroll
    for (int ti = 0; ti < 2; ++ti) {
      const int row = wm * 32 + ti * 16 + lane15;
      u16x8 lo = *(const u16x8*)(&As[p][row * LDS_S + kA]);
      u16x8 hi = *(const u16x8*)(&As[p][row * LDS_S + kA + 16]);
      #pragma unroll
      for (int e = 0; e < 8; ++e) { afrag[ti][e] = lo[e]; afrag[ti][8 + e] = hi[e]; }
    }
    #pragma unroll
    for (int tj = 0; tj < 4; ++tj) {
      const int nrow = wn * 64 + tj * 16 + lane15;
      u16x8 lo = *(const u16x8*)(&Ws[p][nrow * LDS_S + kB]);
      u16x8 hi = *(const u16x8*)(&Ws[p][nrow * LDS_S + kB + 16]);
      #pragma unroll
      for (int e = 0; e < 8; ++e) { bfrag[tj][e] = lo[e]; bfrag[tj][8 + e] = hi[e]; }
    }
    #pragma unroll
    for (int ti = 0; ti < 2; ++ti)
      #pragma unroll
      for (int tj = 0; tj < 4; ++tj)
        acc[ti][tj] = __builtin_amdgcn_wmma_f32_16x16x32_bf16(
            false, __builtin_bit_cast(bf16x16, afrag[ti]),
            false, __builtin_bit_cast(bf16x16, bfrag[tj]),
            (short)0, acc[ti][tj], false, false);
  };

#if USE_ASYNC_LDS
  // ---- async global->LDS double-buffered pipeline (ASYNCcnt) ----
  auto stage_async = [&](int k0, int p) {
    #pragma unroll
    for (int i = 0; i < 2; ++i) {
      async_copy16(aptr[i] + k0, &As[p][srow[i] * LDS_S + skc[i]]);
      async_copy16(wptr[i] + k0, &Ws[p][srow[i] * LDS_S + skc[i]]);
    }
  };
  stage_async(0, 0);
  for (int t = 0; t < T; ++t) {
    const int cur = t & 1;
    if (t + 1 < T) { stage_async((t + 1) * BK, cur ^ 1); s_wait_async<4>(); }
    else           { s_wait_async<0>(); }
    __syncthreads();              // all waves' async writes for tile t landed
    compute(cur);
    __syncthreads();              // done reading before buffer is overwritten
  }
#else
  // ---- register-staged double-buffered pipeline ----
  u16x8 ra[2], rw[2];
  auto gload = [&](int k0) {
    #pragma unroll
    for (int i = 0; i < 2; ++i) {
      ra[i] = *(const u16x8*)(aptr[i] + k0);
      rw[i] = *(const u16x8*)(wptr[i] + k0);
      if (k0 + BK < K) {
        __builtin_prefetch(aptr[i] + k0 + BK, 0, 1);
        __builtin_prefetch(wptr[i] + k0 + BK, 0, 1);
      }
    }
  };
  auto sstore = [&](int p) {
    #pragma unroll
    for (int i = 0; i < 2; ++i) {
      *(u16x8*)(&As[p][srow[i] * LDS_S + skc[i]]) = ra[i];
      *(u16x8*)(&Ws[p][srow[i] * LDS_S + skc[i]]) = rw[i];
    }
  };
  gload(0);
  sstore(0);
  __syncthreads();
  for (int t = 0; t < T; ++t) {
    const int cur = t & 1;
    if (t + 1 < T) gload((t + 1) * BK);
    compute(cur);
    __syncthreads();
    if (t + 1 < T) sstore(cur ^ 1);
    __syncthreads();
  }
#endif

  // ---- epilogue: bias (+relu), bounds-checked store ----
  #pragma unroll
  for (int ti = 0; ti < 2; ++ti) {
    #pragma unroll
    for (int tj = 0; tj < 4; ++tj) {
      const int col = n0 + wn * 64 + tj * 16 + lane15;
      if (col >= N) continue;
      const float bv = bias ? bias[col] : 0.0f;
      #pragma unroll
      for (int e = 0; e < 8; ++e) {
        const int row = m0 + wm * 32 + ti * 16 + ((lane < 16) ? e : 8 + e);
        if (row >= M) continue;
        float v = acc[ti][tj][e] + bv;
        if (RELU) v = v > 0.f ? v : 0.f;
        storeC(C + (size_t)row * ldc + col, v);
      }
    }
  }
}

// =====================================================================
// small kernels
// =====================================================================
__global__ void cvt_f32_bf16_kernel(const float* __restrict__ in,
                                    unsigned short* __restrict__ out, size_t n) {
  for (size_t i = (size_t)blockIdx.x * blockDim.x + threadIdx.x; i < n;
       i += (size_t)gridDim.x * blockDim.x)
    out[i] = f2bf(in[i]);
}

// x[b,l,h] = emb[src[b,l,h/256]][h%256] + pos[l,h]; also masks
__global__ void embed_kernel(const int* __restrict__ src, const float* __restrict__ emb,
                             const float* __restrict__ pos,
                             float* __restrict__ xf, unsigned short* __restrict__ xb,
                             unsigned char* __restrict__ maskp, unsigned char* __restrict__ smask) {
  const size_t idx = (size_t)blockIdx.x * blockDim.x + threadIdx.x;
  if (idx >= (size_t)BB * LL * HH) return;
  const int h = (int)(idx % HH);
  const int bl = (int)(idx / HH);
  const int l = bl % LL, b = bl / LL;
  const int c = h >> 8, d = h & 255;
  const int tok = src[bl * 3 + c];
  const float v = emb[(size_t)tok * 256 + d] + pos[(size_t)l * HH + h];
  xf[idx] = v; xb[idx] = f2bf(v);
  if (h == 0) {
    const unsigned char mp = (src[bl * 3] == 0) ? 1 : 0;
    maskp[bl] = mp;
    smask[b * SSTAR + 1 + l] = mp;
    if (l == 0) smask[b * SSTAR] = 0;
  }
}

__global__ void relay_init_kernel(const float* __restrict__ x,
                                  float* __restrict__ rf, unsigned short* __restrict__ rb) {
  const int idx = blockIdx.x * blockDim.x + threadIdx.x;
  if (idx >= BB * HH) return;
  const int h = idx % HH, b = idx / HH;
  float s = 0.f;
  for (int l = 0; l < LL; ++l) s += x[((size_t)b * LL + l) * HH + h];
  s *= (1.0f / LL);
  rf[idx] = s; rb[idx] = f2bf(s);
}

// row-wise LayerNorm over HH=768 (3 elems/thread, 256 threads/row)
template<bool HAS_RESID, bool LEAKY>
__global__ __launch_bounds__(256)
void ln_kernel(const float* __restrict__ inp, const float* __restrict__ resid,
               const float* __restrict__ g, const float* __restrict__ bta,
               float* __restrict__ out_f, unsigned short* __restrict__ out_b) {
  const int row = blockIdx.x;
  const size_t base = (size_t)row * HH;
  const int tid = threadIdx.x;
  __shared__ float red[8];
  __shared__ float bc;

  float v[3]; float s = 0.f;
  #pragma unroll
  for (int i = 0; i < 3; ++i) {
    const int c = tid + i * 256;
    float x = inp[base + c];
    if (HAS_RESID) x += resid[base + c];
    v[i] = x; s += x;
  }
  #pragma unroll
  for (int o = 16; o > 0; o >>= 1) s += __shfl_xor(s, o, 32);
  if ((tid & 31) == 0) red[tid >> 5] = s;
  __syncthreads();
  if (tid == 0) { float t = 0; for (int i = 0; i < 8; ++i) t += red[i]; bc = t * (1.0f / HH); }
  __syncthreads();
  const float mu = bc;
  float q = 0.f;
  #pragma unroll
  for (int i = 0; i < 3; ++i) { const float d = v[i] - mu; q += d * d; }
  #pragma unroll
  for (int o = 16; o > 0; o >>= 1) q += __shfl_xor(q, o, 32);
  __syncthreads();
  if ((tid & 31) == 0) red[tid >> 5] = q;
  __syncthreads();
  if (tid == 0) { float t = 0; for (int i = 0; i < 8; ++i) t += red[i]; bc = rsqrtf(t * (1.0f / HH) + 1e-5f); }
  __syncthreads();
  const float rstd = bc;
  #pragma unroll
  for (int i = 0; i < 3; ++i) {
    const int c = tid + i * 256;
    float y = (v[i] - mu) * rstd * g[c] + bta[c];
    if (LEAKY) y = (y > 0.f) ? y : LEAKY_SLOPE * y;
    out_f[base + c] = y;
    out_b[base + c] = f2bf(y);
  }
}

// ring attention: one wave per (b,l,head); 4 keys (l-1,l,l+1,relay), zero-padded ends
__global__ __launch_bounds__(256)
void ring_attn_kernel(const float* __restrict__ q, const float* __restrict__ k,
                      const float* __restrict__ v, const float* __restrict__ rk,
                      const float* __restrict__ rv, unsigned short* __restrict__ att_b) {
  const int gt = blockIdx.x * blockDim.x + threadIdx.x;
  const int w = gt >> 5, lane = gt & 31;
  const int head = w % NHEAD;
  const int l = (w / NHEAD) % LL;
  const int b = w / (NHEAD * LL);
  if (b >= BB) return;
  const size_t rowbase = ((size_t)(b * LL + l)) * HH + head * HDIM;
  const float q0 = q[rowbase + lane];
  const float q1 = q[rowbase + lane + 32];

  auto dot64 = [&](const float* p) -> float {
    float s = q0 * p[lane] + q1 * p[lane + 32];
    #pragma unroll
    for (int o = 16; o > 0; o >>= 1) s += __shfl_xor(s, o, 32);
    return s;
  };

  const bool hasL = (l > 0), hasR = (l < LL - 1);
  float s0 = hasL ? dot64(k + rowbase - HH) : 0.f;   // zero-padded key -> score 0, still in softmax
  float s1 = dot64(k + rowbase);
  float s2 = hasR ? dot64(k + rowbase + HH) : 0.f;
  float s3 = dot64(rk + (size_t)b * HH + head * HDIM);
  s0 *= SCALE_ATT; s1 *= SCALE_ATT; s2 *= SCALE_ATT; s3 *= SCALE_ATT;
  const float m = fmaxf(fmaxf(s0, s1), fmaxf(s2, s3));
  float e0 = __expf(s0 - m), e1 = __expf(s1 - m), e2 = __expf(s2 - m), e3 = __expf(s3 - m);
  const float inv = 1.0f / (e0 + e1 + e2 + e3);
  e0 *= inv; e1 *= inv; e2 *= inv; e3 *= inv;

  const float* vc = v + rowbase;
  const float* rvp = rv + (size_t)b * HH + head * HDIM;
  float o0 = e1 * vc[lane]      + e3 * rvp[lane];
  float o1 = e1 * vc[lane + 32] + e3 * rvp[lane + 32];
  if (hasL) { o0 += e0 * vc[(int)lane - HH]; o1 += e0 * vc[lane + 32 - HH]; }
  if (hasR) { o0 += e2 * vc[lane + HH];      o1 += e2 * vc[lane + 32 + HH]; }
  att_b[rowbase + lane]      = f2bf(o0);
  att_b[rowbase + lane + 32] = f2bf(o1);
}

// star attention: one block per (b,head); softmax over 1025 keys with mask
__global__ __launch_bounds__(256)
void star_attn_kernel(const float* __restrict__ q, const float* __restrict__ k,
                      const float* __restrict__ v, const unsigned char* __restrict__ smask,
                      unsigned short* __restrict__ att_b) {
  const int b = blockIdx.x / NHEAD;
  const int head = blockIdx.x % NHEAD;
  const int tid = threadIdx.x;
  __shared__ float p[SSTAR];
  __shared__ float red[8];
  __shared__ float bc;

  const float* qp = q + (size_t)b * HH + head * HDIM;
  float lmax = -3.0e38f;
  for (int s = tid; s < SSTAR; s += 256) {
    const float* kp = k + ((size_t)b * SSTAR + s) * HH + head * HDIM;
    float acc = 0.f;
    #pragma unroll 8
    for (int d = 0; d < HDIM; ++d) acc += qp[d] * kp[d];
    acc *= SCALE_ATT;
    if (smask[b * SSTAR + s]) acc = -3.0e38f;
    p[s] = acc;
    lmax = fmaxf(lmax, acc);
  }
  #pragma unroll
  for (int o = 16; o > 0; o >>= 1) lmax = fmaxf(lmax, __shfl_xor(lmax, o, 32));
  if ((tid & 31) == 0) red[tid >> 5] = lmax;
  __syncthreads();
  if (tid == 0) { float m = red[0]; for (int i = 1; i < 8; ++i) m = fmaxf(m, red[i]); bc = m; }
  __syncthreads();
  const float mx = bc;
  float lsum = 0.f;
  for (int s = tid; s < SSTAR; s += 256) { const float e = __expf(p[s] - mx); p[s] = e; lsum += e; }
  #pragma unroll
  for (int o = 16; o > 0; o >>= 1) lsum += __shfl_xor(lsum, o, 32);
  __syncthreads();
  if ((tid & 31) == 0) red[tid >> 5] = lsum;
  __syncthreads();
  if (tid == 0) { float t = 0; for (int i = 0; i < 8; ++i) t += red[i]; bc = t; }
  __syncthreads();
  const float denom = bc;

  if (tid < HDIM) {
    const float* vp = v + (size_t)b * SSTAR * HH + head * HDIM + tid;
    float acc = 0.f;
    for (int s = 0; s < SSTAR; ++s) acc += p[s] * vp[(size_t)s * HH];
    att_b[(size_t)b * HH + head * HDIM + tid] = f2bf(acc / denom);
  }
}

// y[b,0,:] = relay ; y[b,1+l,:] = nodes[b,l,:]  (bf16)
__global__ void build_y_kernel(const float* __restrict__ relay,
                               const unsigned short* __restrict__ nodes_b,
                               unsigned short* __restrict__ y) {
  const size_t idx = (size_t)blockIdx.x * blockDim.x + threadIdx.x;
  if (idx >= (size_t)BB * SSTAR * HH) return;
  const int h = (int)(idx % HH);
  const int bs = (int)(idx / HH);
  const int s = bs % SSTAR, b = bs / SSTAR;
  y[idx] = (s == 0) ? f2bf(relay[(size_t)b * HH + h])
                    : nodes_b[((size_t)b * LL + (s - 1)) * HH + h];
}

__global__ void leaky_relay_kernel(const float* __restrict__ in,
                                   float* __restrict__ rf, unsigned short* __restrict__ rb) {
  const int idx = blockIdx.x * blockDim.x + threadIdx.x;
  if (idx >= BB * HH) return;
  float v = in[idx];
  v = (v > 0.f) ? v : LEAKY_SLOPE * v;
  rf[idx] = v; rb[idx] = f2bf(v);
}

__global__ void mask_nodes_kernel(float* __restrict__ nf, unsigned short* __restrict__ nb,
                                  const unsigned char* __restrict__ maskp) {
  const size_t idx = (size_t)blockIdx.x * blockDim.x + threadIdx.x;
  if (idx >= (size_t)BB * LL * HH) return;
  if (maskp[idx / HH]) { nf[idx] = 0.f; nb[idx] = 0; }
}

__global__ void build_ft_kernel(const float* __restrict__ relay, const float* __restrict__ nodes,
                                const int* __restrict__ positions, unsigned short* __restrict__ ft) {
  const int idx = blockIdx.x * blockDim.x + threadIdx.x;
  if (idx >= BB * 2 * HH) return;
  const int c = idx % (2 * HH), b = idx / (2 * HH);
  const float v = (c < HH) ? relay[(size_t)b * HH + c]
                           : nodes[((size_t)b * LL + positions[b]) * HH + (c - HH)];
  ft[idx] = f2bf(v);
}

// =====================================================================
// host orchestration
// =====================================================================
extern "C" void kernel_launch(void* const* d_in, const int* in_sizes, int n_in,
                              void* d_out, int out_size, void* d_ws, size_t ws_size,
                              hipStream_t stream) {
  (void)in_sizes; (void)n_in; (void)out_size; (void)ws_size;
  const int*   src_seq  = (const int*)  d_in[0];
  const int*   positions= (const int*)  d_in[1];
  const float* emb      = (const float*)d_in[2];
  const float* pos_tab  = (const float*)d_in[3];
  const float* norm_g   = (const float*)d_in[4];
  const float* norm_b   = (const float*)d_in[5];
  const float* pw_w1    = (const float*)d_in[6];
  const float* pw_b1    = (const float*)d_in[7];
  const float* pw_w2    = (const float*)d_in[8];
  const float* pw_b2    = (const float*)d_in[9];
  const float* pw_g     = (const float*)d_in[10];
  const float* pw_bn    = (const float*)d_in[11];
  const float* ring_wq  = (const float*)d_in[12];
  const float* ring_bq  = (const float*)d_in[13];
  const float* ring_wk  = (const float*)d_in[14];
  const float* ring_bk  = (const float*)d_in[15];
  const float* ring_wv  = (const float*)d_in[16];
  const float* ring_bv  = (const float*)d_in[17];
  const float* ring_wo  = (const float*)d_in[18];
  const float* ring_bo  = (const float*)d_in[19];
  const float* star_wq  = (const float*)d_in[20];
  const float* star_bq  = (const float*)d_in[21];
  const float* star_wk  = (const float*)d_in[22];
  const float* star_bk  = (const float*)d_in[23];
  const float* star_wv  = (const float*)d_in[24];
  const float* star_bv  = (const float*)d_in[25];
  const float* star_wo  = (const float*)d_in[26];
  const float* star_bo  = (const float*)d_in[27];
  const float* head_w1  = (const float*)d_in[28];
  const float* head_b1  = (const float*)d_in[29];
  const float* head_w2  = (const float*)d_in[30];
  const float* head_b2  = (const float*)d_in[31];
  float* out = (float*)d_out;

  // bump allocator over d_ws
  char* ws = (char*)d_ws;
  size_t off = 0;
  auto alloc = [&](size_t bytes) -> void* {
    void* p = (void*)(ws + off);
    off += (bytes + 255) & ~(size_t)255;
    return p;
  };

  // bf16 weight copies
  unsigned short* pw1_b = (unsigned short*)alloc((size_t)NLAYERS*DINNER*HH*2);
  unsigned short* pw2_b = (unsigned short*)alloc((size_t)NLAYERS*HH*DINNER*2);
  unsigned short* rwq_b = (unsigned short*)alloc((size_t)NLAYERS*HH*HH*2);
  unsigned short* rwk_b = (unsigned short*)alloc((size_t)NLAYERS*HH*HH*2);
  unsigned short* rwv_b = (unsigned short*)alloc((size_t)NLAYERS*HH*HH*2);
  unsigned short* rwo_b = (unsigned short*)alloc((size_t)NLAYERS*HH*HH*2);
  unsigned short* swq_b = (unsigned short*)alloc((size_t)NLAYERS*HH*HH*2);
  unsigned short* swk_b = (unsigned short*)alloc((size_t)NLAYERS*HH*HH*2);
  unsigned short* swv_b = (unsigned short*)alloc((size_t)NLAYERS*HH*HH*2);
  unsigned short* swo_b = (unsigned short*)alloc((size_t)NLAYERS*HH*HH*2);
  unsigned short* hw1_b = (unsigned short*)alloc((size_t)3*NHID*2*HH*2);
  unsigned short* hw2_b = (unsigned short*)alloc((size_t)3*NOUT*NHID*2);
  // activations
  float*          nodes_f = (float*)alloc((size_t)BB*LL*HH*4);
  unsigned short* nodes_b = (unsigned short*)alloc((size_t)BB*LL*HH*2);
  float*          relay_f = (float*)alloc((size_t)BB*HH*4);
  unsigned short* relay_b = (unsigned short*)alloc((size_t)BB*HH*2);
  unsigned char*  maskp   = (unsigned char*)alloc((size_t)BB*LL);
  unsigned char*  smask   = (unsigned char*)alloc((size_t)BB*SSTAR);
  unsigned short* h1_b    = (unsigned short*)alloc((size_t)BB*LL*DINNER*2);
  float*          h2_f    = (float*)alloc((size_t)BB*LL*HH*4);
  float*          h_f     = (float*)alloc((size_t)BB*LL*HH*4);
  unsigned short* h_b     = (unsigned short*)alloc((size_t)BB*LL*HH*2);
  float*          q_f     = (float*)alloc((size_t)BB*LL*HH*4);
  float*          k_f     = (float*)alloc((size_t)BB*LL*HH*4);
  float*          v_f     = (float*)alloc((size_t)BB*LL*HH*4);
  float*          rk_f    = (float*)alloc((size_t)BB*HH*4);
  float*          rv_f    = (float*)alloc((size_t)BB*HH*4);
  unsigned short* att_b   = (unsigned short*)alloc((size_t)BB*LL*HH*2);
  float*          ringo_f = (float*)alloc((size_t)BB*LL*HH*4);
  unsigned short* y_b     = (unsigned short*)alloc((size_t)BB*SSTAR*HH*2);
  float*          qs_f    = (float*)alloc((size_t)BB*HH*4);
  float*          ks_f    = (float*)alloc((size_t)BB*SSTAR*HH*4);
  float*          vs_f    = (float*)alloc((size_t)BB*SSTAR*HH*4);
  unsigned short* satt_b  = (unsigned short*)alloc((size_t)BB*HH*2);
  float*          staro_f = (float*)alloc((size_t)BB*HH*4);
  unsigned short* ft_b    = (unsigned short*)alloc((size_t)BB*2*HH*2);
  unsigned short* hid_b   = (unsigned short*)alloc((size_t)BB*NHID*2);

  auto cvt = [&](const float* src, unsigned short* dst, size_t n) {
    cvt_f32_bf16_kernel<<<dim3(4096), 256, 0, stream>>>(src, dst, n);
  };
  // weight conversions (every call; deterministic)
  cvt(pw_w1,  pw1_b, (size_t)NLAYERS*DINNER*HH);
  cvt(pw_w2,  pw2_b, (size_t)NLAYERS*HH*DINNER);
  cvt(ring_wq, rwq_b, (size_t)NLAYERS*HH*HH);
  cvt(ring_wk, rwk_b, (size_t)NLAYERS*HH*HH);
  cvt(ring_wv, rwv_b, (size_t)NLAYERS*HH*HH);
  cvt(ring_wo, rwo_b, (size_t)NLAYERS*HH*HH);
  cvt(star_wq, swq_b, (size_t)NLAYERS*HH*HH);
  cvt(star_wk, swk_b, (size_t)NLAYERS*HH*HH);
  cvt(star_wv, swv_b, (size_t)NLAYERS*HH*HH);
  cvt(star_wo, swo_b, (size_t)NLAYERS*HH*HH);
  cvt(head_w1, hw1_b, (size_t)3*NHID*2*HH);
  cvt(head_w2, hw2_b, (size_t)3*NOUT*NHID);

  auto gemmF = [&](const unsigned short* A, const unsigned short* W, const float* bias,
                   float* C, int M, int N, int K, int ldc) {
    dim3 g((N + 127) / 128, (M + 127) / 128);
    gemm_bf16_wmma<false, float><<<g, 256, 0, stream>>>(A, W, bias, C, M, N, K, ldc);
  };
  auto gemmBrelu = [&](const unsigned short* A, const unsigned short* W, const float* bias,
                       unsigned short* C, int M, int N, int K, int ldc) {
    dim3 g((N + 127) / 128, (M + 127) / 128);
    gemm_bf16_wmma<true, unsigned short><<<g, 256, 0, stream>>>(A, W, bias, C, M, N, K, ldc);
  };

  const int TOK = BB * LL;                       // 8192
  // embedding + positional encoding + masks
  {
    const size_t n = (size_t)TOK * HH;
    embed_kernel<<<dim3((unsigned)((n + 255) / 256)), 256, 0, stream>>>(
        src_seq, emb, pos_tab, nodes_f, nodes_b, maskp, smask);
  }
  relay_init_kernel<<<dim3((BB * HH + 255) / 256), 256, 0, stream>>>(nodes_f, relay_f, relay_b);

  for (int i = 0; i < NLAYERS; ++i) {
    const unsigned short* W1 = pw1_b + (size_t)i * DINNER * HH;
    const unsigned short* W2 = pw2_b + (size_t)i * HH * DINNER;
    const unsigned short* Wq = rwq_b + (size_t)i * HH * HH;
    const unsigned short* Wk = rwk_b + (size_t)i * HH * HH;
    const unsigned short* Wv = rwv_b + (size_t)i * HH * HH;
    const unsigned short* Wo = rwo_b + (size_t)i * HH * HH;
    const unsigned short* Sq = swq_b + (size_t)i * HH * HH;
    const unsigned short* Sk = swk_b + (size_t)i * HH * HH;
    const unsigned short* Sv = swv_b + (size_t)i * HH * HH;
    const unsigned short* So = swo_b + (size_t)i * HH * HH;

    // ---- PWFF: h = LN(relu(nodes*W1^T+b1)*W2^T+b2 + nodes) ----
    gemmBrelu(nodes_b, W1, pw_b1 + i * DINNER, h1_b, TOK, DINNER, HH, DINNER);
    gemmF(h1_b, W2, pw_b2 + i * HH, h2_f, TOK, HH, DINNER, HH);
    ln_kernel<true, false><<<dim3(TOK), 256, 0, stream>>>(
        h2_f, nodes_f, pw_g + i * HH, pw_bn + i * HH, h_f, h_b);

    // ---- ring attention (msa1) ----
    gemmF(h_b,     Wq, ring_bq + i * HH, q_f,  TOK, HH, HH, HH);
    gemmF(h_b,     Wk, ring_bk + i * HH, k_f,  TOK, HH, HH, HH);
    gemmF(h_b,     Wv, ring_bv + i * HH, v_f,  TOK, HH, HH, HH);
    gemmF(relay_b, Wk, ring_bk + i * HH, rk_f, BB,  HH, HH, HH);
    gemmF(relay_b, Wv, ring_bv + i * HH, rv_f, BB,  HH, HH, HH);
    ring_attn_kernel<<<dim3(TOK * NHEAD / 8), 256, 0, stream>>>(q_f, k_f, v_f, rk_f, rv_f, att_b);
    gemmF(att_b, Wo, ring_bo + i * HH, ringo_f, TOK, HH, HH, HH);
    ln_kernel<false, true><<<dim3(TOK), 256, 0, stream>>>(
        ringo_f, nullptr, norm_g + i * HH, norm_b + i * HH, nodes_f, nodes_b);

    // ---- star attention (msa2) ----
    {
      const size_t n = (size_t)BB * SSTAR * HH;
      build_y_kernel<<<dim3((unsigned)((n + 255) / 256)), 256, 0, stream>>>(relay_f, nodes_b, y_b);
    }
    gemmF(relay_b, Sq, star_bq + i * HH, qs_f, BB,         HH, HH, HH);
    gemmF(y_b,     Sk, star_bk + i * HH, ks_f, BB * SSTAR, HH, HH, HH);
    gemmF(y_b,     Sv, star_bv + i * HH, vs_f, BB * SSTAR, HH, HH, HH);
    star_attn_kernel<<<dim3(BB * NHEAD), 256, 0, stream>>>(qs_f, ks_f, vs_f, smask, satt_b);
    gemmF(satt_b, So, star_bo + i * HH, staro_f, BB, HH, HH, HH);
    leaky_relay_kernel<<<dim3((BB * HH + 255) / 256), 256, 0, stream>>>(staro_f, relay_f, relay_b);

    // ---- zero padded node rows ----
    {
      const size_t n = (size_t)TOK * HH;
      mask_nodes_kernel<<<dim3((unsigned)((n + 255) / 256)), 256, 0, stream>>>(nodes_f, nodes_b, maskp);
    }
  }

  // ---- heads ----
  build_ft_kernel<<<dim3((BB * 2 * HH + 255) / 256), 256, 0, stream>>>(relay_f, nodes_f, positions, ft_b);
  for (int j = 0; j < 3; ++j) {
    gemmBrelu(ft_b, hw1_b + (size_t)j * NHID * 2 * HH, head_b1 + j * NHID, hid_b,
              BB, NHID, 2 * HH, NHID);
    gemmF(hid_b, hw2_b + (size_t)j * NOUT * NHID, head_b2 + j * NOUT, out + (size_t)j * NOUT,
          BB, NOUT, NHID, 3 * NOUT);
  }
}
